// ADMMNet_77532749627571
// MI455X (gfx1250) — matrix operand order repaired and checked by reading, not compiled, not run on hardware
//
#include <hip/hip_runtime.h>
#include <cstdint>

#define S_STAGES 12
#define BATCH    8192
#define NDIM     1024

typedef __bf16 bf16_t;
typedef __attribute__((ext_vector_type(4)))  __bf16 v4bf;
typedef __attribute__((ext_vector_type(8)))  __bf16 v8bf;
typedef __attribute__((ext_vector_type(16))) __bf16 v16bf;
typedef __attribute__((ext_vector_type(8)))  float  v8f;

// ---------------------------------------------------------------------------
// CDNA5 helpers
// ---------------------------------------------------------------------------
__device__ __forceinline__ v8f wmma_bf16(v16bf a, v16bf b, v8f c) {
  // v_wmma_f32_16x16x32_bf16  D = A(16x32) * B(32x16) + C
  return __builtin_amdgcn_wmma_f32_16x16x32_bf16(
      /*neg_a=*/false, a, /*neg_b=*/false, b,
      /*c_mod=*/(short)0, c, /*reuse_a=*/false, /*reuse_b=*/false);
}

// Async DMA: 16 bytes per lane, global -> LDS, tracked by ASYNCcnt.
__device__ __forceinline__ void async_b128(uint32_t lds_off, const void* gptr) {
  asm volatile("global_load_async_to_lds_b128 %0, %1, off"
               :: "v"(lds_off), "v"(gptr) : "memory");
}

// Retire all but the 8 prefetch ops still in flight (double-buffer pipeline).
__device__ __forceinline__ void wait_async_le8() {
  asm volatile("s_wait_asynccnt 0x8" ::: "memory");
}
__device__ __forceinline__ void wait_async_le0() {
  asm volatile("s_wait_asynccnt 0x0" ::: "memory");
}

__device__ __forceinline__ uint32_t lds_addr(const void* p) {
  return (uint32_t)(uintptr_t)p;  // low 32 bits of generic ptr == LDS offset
}

// ---------------------------------------------------------------------------
// One-time: split fp32 M2[S,N,N] into bf16 hi + bf16 residual lo
// ---------------------------------------------------------------------------
__global__ __launch_bounds__(256) void k_split(const float* __restrict__ src,
                                               bf16_t* __restrict__ hi,
                                               bf16_t* __restrict__ lo) {
  const size_t i = ((size_t)blockIdx.x * 256 + threadIdx.x) * 4;
  float4 v = *(const float4*)(src + i);
  float t[4] = {v.x, v.y, v.z, v.w};
  v4bf hv, lv;
#pragma unroll
  for (int j = 0; j < 4; ++j) {
    bf16_t h = (bf16_t)t[j];
    hv[j] = h;
    lv[j] = (bf16_t)(t[j] - (float)h);
  }
  *(v4bf*)(hi + i) = hv;
  *(v4bf*)(lo + i) = lv;
}

__global__ __launch_bounds__(256) void k_zero(float* __restrict__ p) {
  const size_t i = ((size_t)blockIdx.x * 256 + threadIdx.x) * 4;
  *(float4*)(p + i) = make_float4(0.f, 0.f, 0.f, 0.f);
}

// ---------------------------------------------------------------------------
// Per-stage: t = y + rho*(z - u), split into bf16 hi/lo   (all [B,N])
// ---------------------------------------------------------------------------
__global__ __launch_bounds__(256) void k_prepare(const float* __restrict__ y,
                                                 const float* __restrict__ z,
                                                 const float* __restrict__ u,
                                                 const float* __restrict__ rho,
                                                 bf16_t* __restrict__ thi,
                                                 bf16_t* __restrict__ tlo) {
  const size_t i = ((size_t)blockIdx.x * 256 + threadIdx.x) * 4;
  const float r = rho[0];
  float4 yv = *(const float4*)(y + i);
  float4 zv = *(const float4*)(z + i);
  float4 uv = *(const float4*)(u + i);
  float t[4] = {yv.x + r * (zv.x - uv.x), yv.y + r * (zv.y - uv.y),
                yv.z + r * (zv.z - uv.z), yv.w + r * (zv.w - uv.w)};
  v4bf hv, lv;
#pragma unroll
  for (int j = 0; j < 4; ++j) {
    bf16_t h = (bf16_t)t[j];
    hv[j] = h;
    lv[j] = (bf16_t)(t[j] - (float)h);
  }
  *(v4bf*)(thi + i) = hv;
  *(v4bf*)(tlo + i) = lv;
}

// ---------------------------------------------------------------------------
// Stage GEMM + fused ADMM epilogue.
//   C[b,n] = sum_k T[b,k] * M2[n,k]      (i.e. C = T @ M2^T, batch-major)
// bf16x3: C = Thi@Bhi + Thi@Blo + Tlo@Bhi, fp32 accumulate.
// Workgroup tile 128(M=batch) x 128(N), K-step 32, 8 waves of 32 lanes.
// Double-buffered LDS; async DMA of tile k+1 overlaps WMMA on tile k.
// ---------------------------------------------------------------------------
__global__ __launch_bounds__(256) void k_stage_gemm(
    const bf16_t* __restrict__ thi, const bf16_t* __restrict__ tlo,
    const bf16_t* __restrict__ bhi, const bf16_t* __restrict__ blo,
    const float* __restrict__ m3s,
    const float* __restrict__ alpha, const float* __restrict__ lamb,
    const float* __restrict__ rho,
    float* __restrict__ zbuf, float* __restrict__ ubuf, int stage) {
  __shared__ bf16_t sAhi[2][128 * 32];
  __shared__ bf16_t sAlo[2][128 * 32];
  __shared__ bf16_t sBhi[2][128 * 32];
  __shared__ bf16_t sBlo[2][128 * 32];

  const int tid  = threadIdx.x;
  const int lane = tid & 31;
  const int wave = tid >> 5;   // 0..7
  const int wm   = wave & 3;   // 4 row-blocks of 32
  const int wn   = wave >> 2;  // 2 col-blocks of 64
  const int l16  = lane & 15;

  const int blockRow = blockIdx.y * 128;  // over BATCH
  const int blockCol = blockIdx.x * 128;  // over NDIM

  v8f acc[2][4] = {};

  // Fragment K-phases chosen here (contiguous k order, same for A and B, wave32):
  //   A 16x32 bf16: lanes 0-15 hold K = [0..15]; lanes 16-31: K = [16..31]
  //   B 32x16 bf16: lanes 0-15 hold K = [0..15]; lanes 16-31: K = [16..31]
  const int cA = (lane < 16) ? 0 : 16;
  const int cB = (lane < 16) ? 0 : 16;

  // Issue the 8 async b128 DMAs (this thread's share of one 4-buffer tile).
  auto issue_tile = [&](int buf, int kk) {
#pragma unroll
    for (int c = 0; c < 2; ++c) {
      const int chunk = tid + c * 256;   // 0..511
      const int row   = chunk >> 2;      // 0..127
      const int off   = (chunk & 3) * 8; // bf16 elements within 32-wide slab
      const size_t ga = (size_t)(blockRow + row) * NDIM + kk + off;
      const size_t gb = (size_t)(blockCol + row) * NDIM + kk + off;
      const int ls = row * 32 + off;
      async_b128(lds_addr(&sAhi[buf][ls]), (const void*)(thi + ga));
      async_b128(lds_addr(&sAlo[buf][ls]), (const void*)(tlo + ga));
      async_b128(lds_addr(&sBhi[buf][ls]), (const void*)(bhi + gb));
      async_b128(lds_addr(&sBlo[buf][ls]), (const void*)(blo + gb));
    }
  };

  issue_tile(0, 0);

#pragma unroll 2
  for (int kk = 0; kk < NDIM; kk += 32) {
    const int cur  = (kk >> 5) & 1;
    const bool more = (kk + 32) < NDIM;
    if (more) {
      issue_tile(cur ^ 1, kk + 32);  // prefetch next tile (8 ops in flight)
      wait_async_le8();              // retire current tile's 8 DMAs only
    } else {
      wait_async_le0();
    }
    __syncthreads();                 // current tile visible to all waves

    // --- fragments from LDS
    v16bf aHi[2], aLo[2], bHi[4], bLo[4];
#pragma unroll
    for (int mi = 0; mi < 2; ++mi) {
      const bf16_t* rp = &sAhi[cur][(wm * 32 + mi * 16 + l16) * 32];
      const bf16_t* rq = &sAlo[cur][(wm * 32 + mi * 16 + l16) * 32];
      v8bf p0 = *(const v8bf*)(rp + cA);
      v8bf p1 = *(const v8bf*)(rp + cA + 8);
      aHi[mi] = __builtin_shufflevector(p0, p1, 0, 1, 2, 3, 4, 5, 6, 7, 8, 9,
                                        10, 11, 12, 13, 14, 15);
      v8bf q0 = *(const v8bf*)(rq + cA);
      v8bf q1 = *(const v8bf*)(rq + cA + 8);
      aLo[mi] = __builtin_shufflevector(q0, q1, 0, 1, 2, 3, 4, 5, 6, 7, 8, 9,
                                        10, 11, 12, 13, 14, 15);
    }
#pragma unroll
    for (int ni = 0; ni < 4; ++ni) {
      bHi[ni] = *(const v16bf*)(&sBhi[cur][(wn * 64 + ni * 16 + l16) * 32 + cB]);
      bLo[ni] = *(const v16bf*)(&sBlo[cur][(wn * 64 + ni * 16 + l16) * 32 + cB]);
    }

    // --- bf16x3 product terms; 8 independent accumulators per term
#pragma unroll
    for (int mi = 0; mi < 2; ++mi)
#pragma unroll
      for (int ni = 0; ni < 4; ++ni)
        acc[mi][ni] = wmma_bf16(aHi[mi], bHi[ni], acc[mi][ni]);
#pragma unroll
    for (int mi = 0; mi < 2; ++mi)
#pragma unroll
      for (int ni = 0; ni < 4; ++ni)
        acc[mi][ni] = wmma_bf16(aHi[mi], bLo[ni], acc[mi][ni]);
#pragma unroll
    for (int mi = 0; mi < 2; ++mi)
#pragma unroll
      for (int ni = 0; ni < 4; ++ni)
        acc[mi][ni] = wmma_bf16(aLo[mi], bHi[ni], acc[mi][ni]);

    __syncthreads();  // done reading cur; next iter may DMA over it
  }

  // --- fused ADMM epilogue (C/D layout: VGPR r, lanes 0-15 -> M=r, N=lane;
  //     lanes 16-31 -> M=r+8, N=lane-16)
  const float aS  = alpha[stage];
  const float thr = lamb[stage] / rho[0];
#pragma unroll
  for (int mi = 0; mi < 2; ++mi) {
#pragma unroll
    for (int ni = 0; ni < 4; ++ni) {
      const int col = blockCol + wn * 64 + ni * 16 + l16;
      const float m3v = m3s[col];
      const int rowBase =
          blockRow + wm * 32 + mi * 16 + ((lane >> 4) << 3);
#pragma unroll
      for (int r = 0; r < 8; ++r) {
        const size_t idx = (size_t)(rowBase + r) * NDIM + col;
        const float zo = zbuf[idx];
        const float uo = ubuf[idx];
        const float x  = m3v * acc[mi][ni][r];
        const float x1 = aS * x + (1.0f - aS) * zo;
        const float v  = x1 + uo;
        float m = fabsf(v) - thr;
        m = m > 0.0f ? m : 0.0f;
        const float z1 = copysignf(m, v);
        zbuf[idx] = z1;
        ubuf[idx] = uo + x1 - z1;
      }
    }
  }
}

// ---------------------------------------------------------------------------
// Host launcher
// ---------------------------------------------------------------------------
extern "C" void kernel_launch(void* const* d_in, const int* in_sizes, int n_in,
                              void* d_out, int out_size, void* d_ws,
                              size_t ws_size, hipStream_t stream) {
  const float* y     = (const float*)d_in[0];  // [B,N]
  const float* M2    = (const float*)d_in[1];  // [S,N,N]
  const float* M3    = (const float*)d_in[2];  // [S,N]
  const float* alpha = (const float*)d_in[3];  // [S]
  const float* lamb  = (const float*)d_in[4];  // [S]
  const float* rho   = (const float*)d_in[5];  // [1]

  char* w = (char*)d_ws;
  bf16_t* m2hi = (bf16_t*)w; w += (size_t)S_STAGES * NDIM * NDIM * sizeof(bf16_t);
  bf16_t* m2lo = (bf16_t*)w; w += (size_t)S_STAGES * NDIM * NDIM * sizeof(bf16_t);
  bf16_t* thi  = (bf16_t*)w; w += (size_t)BATCH * NDIM * sizeof(bf16_t);
  bf16_t* tlo  = (bf16_t*)w; w += (size_t)BATCH * NDIM * sizeof(bf16_t);
  float*  zbuf = (float*)w;  w += (size_t)BATCH * NDIM * sizeof(float);
  float*  ubuf = (float*)w;  w += (size_t)BATCH * NDIM * sizeof(float);

  // one-time per launch: split M2 to bf16 hi/lo; zero z & u (contiguous)
  k_split<<<(S_STAGES * NDIM * NDIM) / 1024, 256, 0, stream>>>(M2, m2hi, m2lo);
  k_zero<<<(2 * BATCH * NDIM) / 1024, 256, 0, stream>>>(zbuf);

  for (int s = 0; s < S_STAGES; ++s) {
    k_prepare<<<(BATCH * NDIM) / 1024, 256, 0, stream>>>(y, zbuf, ubuf, rho,
                                                         thi, tlo);
    dim3 grid(NDIM / 128, BATCH / 128);
    k_stage_gemm<<<grid, 256, 0, stream>>>(
        thi, tlo, m2hi + (size_t)s * NDIM * NDIM,
        m2lo + (size_t)s * NDIM * NDIM, M3 + (size_t)s * NDIM, alpha, lamb,
        rho, zbuf, ubuf, s);
  }

  hipMemcpyAsync(d_out, zbuf, (size_t)BATCH * NDIM * sizeof(float),
                 hipMemcpyDeviceToDevice, stream);
}
